// SwinTransformerBlock_17540646437104
// MI455X (gfx1250) — matrix operand order repaired and checked by reading, not compiled, not run on hardware
//
#include <hip/hip_runtime.h>
#include <hip/hip_bf16.h>
#include <math.h>

typedef __attribute__((ext_vector_type(16))) _Float16 v16h;
typedef __attribute__((ext_vector_type(8)))  _Float16 v8h;
typedef __attribute__((ext_vector_type(8)))  float    v8f;
typedef __attribute__((ext_vector_type(4)))  int      v4i;

#define C_    384
#define CM_   1536
#define NHEAD 12
#define HD    32
#define NTOK  49
#define NWIN  1024
#define LTOK  50176
#define MPAD  65536   /* NWIN * 64 padded rows */

// LDS B-tile: 32 rows x 128 cols f16, padded stride 136 halves (272B = 68 dwords
// -> +4 banks per row -> the 16 k-rows of a B-fragment land in distinct 4-bank
// groups; B-fragment reads are 16-bit LDS loads). 16B alignment preserved.
#define LDSB_STRIDE 136
#define LDSB_ELEMS  (32 * LDSB_STRIDE)

// ---- CDNA5 async global->LDS path (guarded; fallback = ld + ds_store) ----
#if defined(__has_builtin)
#if __has_builtin(__builtin_amdgcn_global_load_async_to_lds_b128) && \
    __has_builtin(__builtin_amdgcn_s_wait_asynccnt)
#define USE_ASYNC_LDS 1
#endif
#endif
#ifndef USE_ASYNC_LDS
#define USE_ASYNC_LDS 0
#endif

union FragU { v16h v; v8h h[2]; };

__device__ inline v8f vzero8() {
  v8f z = {0.f,0.f,0.f,0.f,0.f,0.f,0.f,0.f};
  return z;
}

#define WMMA_F16(A_, B_, C_ACC) \
  __builtin_amdgcn_wmma_f32_16x16x32_f16(false, (A_), false, (B_), (short)0, (C_ACC), false, false)

// ---- fragment loaders (wave32 layouts per CDNA5 ISA 7.12.2) -------------
// A 16x32 f16: lane l -> row m0+(l&15), halves: K = 8*(l>>4)+0..7 and +16..23
__device__ inline v16h load_a_frag(const _Float16* base, int lda, int m0, int k0) {
  int lane = threadIdx.x & 31;
  const _Float16* p = base + (size_t)(m0 + (lane & 15)) * lda + k0 + ((lane >> 4) << 3);
  FragU u; u.h[0] = *(const v8h*)p; u.h[1] = *(const v8h*)(p + 16);
  return u.v;
}
// B 32x16 f16 (K-major memory): lane l -> col n0+(l&15), K = 8*(l>>4)+0..7 and +16..23
__device__ inline v16h load_b_frag(const _Float16* base, int ldb, int k0, int n0) {
  int lane = threadIdx.x & 31;
  const _Float16* p = base + (size_t)(k0 + ((lane >> 4) << 3)) * ldb + n0 + (lane & 15);
  FragU u;
#pragma unroll
  for (int s = 0; s < 16; ++s) u.v[s] = p[(size_t)(s + (s & 8)) * ldb];
  return u.v;
}

// ---- stage one 32x128 f16 B tile into LDS (256 threads, 2 x 16B each) ---
__device__ inline void stage_b_tile(const _Float16* __restrict__ src, int ldb,
                                    _Float16* dst) {
  int t = threadIdx.x;
#pragma unroll
  for (int i = 0; i < 2; ++i) {
    int c = t + i * 256;                 // 512 chunks of 16B
    int row = c >> 4, col = (c & 15) << 3;
    const _Float16* s = src + (size_t)row * ldb + col;
    _Float16* d = dst + row * LDSB_STRIDE + col;
#if USE_ASYNC_LDS
    __builtin_amdgcn_global_load_async_to_lds_b128((v4i*)s, (v4i*)d, 0, 0);
#else
    *(v8h*)d = *(const v8h*)s;           // global_load_b128 + ds_store_b128
#endif
  }
}

__device__ inline void stage_wait_barrier() {
#if USE_ASYNC_LDS
  __builtin_amdgcn_s_wait_asynccnt(0);   // own async copies done
#endif
  __syncthreads();                       // everyone's copies done
}

// ---- GEMM core: block stages B in LDS; one wave computes 32(M)x64(N) ----
// schedule per stage s: wait+barrier (stage s visible, stage s-1 consumers
// done) -> issue stage s+1 into the freed buffer -> consume stage s.
// Fully unrolled (K is a literal at every call site): keeps accumulators
// pinned so WMMAs chain in place with no phi copies / hazard NOPs.
__device__ inline void gemm_core_lds(const _Float16* __restrict__ A, int lda,
                                     const _Float16* __restrict__ B, int ldb,
                                     int K, int m0, int nblk, int n0w,
                                     _Float16* ldsB, v8f acc[2][4]) {
  const int lane = threadIdx.x & 31;
  const int lo = lane & 15, hi = lane >> 4;
  const _Float16* pa0 = A + (size_t)(m0 + lo) * lda + (hi << 3);
  const _Float16* pa1 = pa0 + (size_t)16 * lda;
  const int krow = hi << 3;
  const int nstages = K / 32;
  stage_b_tile(B + nblk, ldb, ldsB);
#pragma unroll
  for (int s = 0; s < nstages; ++s) {
    stage_wait_barrier();
    if (s + 1 < nstages)
      stage_b_tile(B + nblk + (size_t)(s + 1) * 32 * ldb, ldb,
                   ldsB + ((s + 1) & 1) * LDSB_ELEMS);
    __builtin_prefetch(pa0 + 32, 0, 3);
    const _Float16* pbl = ldsB + (s & 1) * LDSB_ELEMS + krow * LDSB_STRIDE + n0w + lo;
    FragU a0, a1, b0, b1, b2, b3;
    a0.h[0] = *(const v8h*)(pa0);       a0.h[1] = *(const v8h*)(pa0 + 16);
    a1.h[0] = *(const v8h*)(pa1);       a1.h[1] = *(const v8h*)(pa1 + 16);
#pragma unroll
    for (int e = 0; e < 16; ++e) {
      const _Float16* q = pbl + (e + (e & 8)) * LDSB_STRIDE;
      b0.v[e] = q[0]; b1.v[e] = q[16]; b2.v[e] = q[32]; b3.v[e] = q[48];
    }
    acc[0][0] = WMMA_F16(a0.v, b0.v, acc[0][0]);
    acc[0][1] = WMMA_F16(a0.v, b1.v, acc[0][1]);
    acc[0][2] = WMMA_F16(a0.v, b2.v, acc[0][2]);
    acc[0][3] = WMMA_F16(a0.v, b3.v, acc[0][3]);
    acc[1][0] = WMMA_F16(a1.v, b0.v, acc[1][0]);
    acc[1][1] = WMMA_F16(a1.v, b1.v, acc[1][1]);
    acc[1][2] = WMMA_F16(a1.v, b2.v, acc[1][2]);
    acc[1][3] = WMMA_F16(a1.v, b3.v, acc[1][3]);
    pa0 += 32; pa1 += 32;
  }
}

// ---- weight transpose + f16: dst (I x O), dst[i*O+o] = src[o*I+i] -------
__global__ __launch_bounds__(256) void k_transpose_w(const float* __restrict__ src,
                                                     _Float16* __restrict__ dst,
                                                     int O, int I) {
  int idx = blockIdx.x * 256 + threadIdx.x;
  if (idx >= O * I) return;
  int i = idx / O, o = idx - i * O;
  dst[idx] = (_Float16)src[(size_t)o * I + i];
}

// ---- combined relative-position bias + shift-window mask table ----------
// table[type(4)][head(12)][m(64)][n(64)] f32; type = 2*(wh==31) + (ww==31)
__global__ __launch_bounds__(256) void k_biastab(const float* __restrict__ rpb,
                                                 float* __restrict__ tab) {
  int idx = blockIdx.x * 256 + threadIdx.x;
  if (idx >= 4 * NHEAD * 64 * 64) return;
  int type = idx / (NHEAD * 64 * 64);
  int rem  = idx - type * (NHEAD * 64 * 64);
  int head = rem / 4096;
  int mn   = rem - head * 4096;
  int m = mn >> 6, n = mn & 63;
  float val;
  if (n >= NTOK)       val = -1e9f;      // padded key columns: softmax -> 0
  else if (m >= NTOK)  val = 0.f;        // padded query rows: never stored
  else {
    int r1 = m / 7, c1 = m - r1 * 7, r2 = n / 7, c2 = n - r2 * 7;
    int ri = (r1 - r2 + 6) * 13 + (c1 - c2 + 6);
    float bias = rpb[(size_t)ri * NHEAD + head];
    int th = type >> 1, tw = type & 1;
    int g1 = (th ? (r1 < 4 ? 1 : 2) : 0) * 3 + (tw ? (c1 < 4 ? 1 : 2) : 0);
    int g2 = (th ? (r2 < 4 ? 1 : 2) : 0) * 3 + (tw ? (c2 < 4 ? 1 : 2) : 0);
    val = bias + ((g1 != g2) ? -100.f : 0.f);
  }
  tab[idx] = val;
}

// ---- LN1 + cyclic shift + window partition -> winA f16 (padded 64 rows) -
__global__ __launch_bounds__(256) void k_ln1_window(const float* __restrict__ x,
    const float* __restrict__ w, const float* __restrict__ b,
    _Float16* __restrict__ winA) {
  int wid = threadIdx.x >> 5, lane = threadIdx.x & 31;
  int row = blockIdx.x * 8 + wid;            // 0..65535
  _Float16* dst = winA + (size_t)row * C_;
  int t = row & 63, win = row >> 6;
  if (t >= NTOK) {                            // zero the padding rows
#pragma unroll
    for (int i = 0; i < 12; ++i) dst[lane + 32 * i] = (_Float16)0.f;
    return;
  }
  int wh = win >> 5, wc = win & 31;
  int r = t / 7, c = t - r * 7;
  int hp = (wh * 7 + r + 3) % 224;            // shifted coord -> source coord
  int wp = (wc * 7 + c + 3) % 224;
  const float* src = x + (size_t)(hp * 224 + wp) * C_;
  float v[12]; float s = 0.f;
#pragma unroll
  for (int i = 0; i < 12; ++i) { v[i] = src[lane + 32 * i]; s += v[i]; }
#pragma unroll
  for (int off = 16; off; off >>= 1) s += __shfl_xor(s, off, 32);
  float mu = s * (1.f / 384.f);
  float var = 0.f;
#pragma unroll
  for (int i = 0; i < 12; ++i) { float d = v[i] - mu; var += d * d; }
#pragma unroll
  for (int off = 16; off; off >>= 1) var += __shfl_xor(var, off, 32);
  float inv = rsqrtf(var * (1.f / 384.f) + 1e-5f);
#pragma unroll
  for (int i = 0; i < 12; ++i)
    dst[lane + 32 * i] = (_Float16)((v[i] - mu) * inv * w[lane + 32 * i] + b[lane + 32 * i]);
}

// ---- LN2 over x1 (flat spatial order) -> y f16 --------------------------
__global__ __launch_bounds__(256) void k_ln2(const float* __restrict__ xin,
    const float* __restrict__ w, const float* __restrict__ b,
    _Float16* __restrict__ y) {
  int wid = threadIdx.x >> 5, lane = threadIdx.x & 31;
  size_t row = (size_t)blockIdx.x * 8 + wid;
  const float* src = xin + row * C_;
  _Float16* dst = y + row * C_;
  float v[12]; float s = 0.f;
#pragma unroll
  for (int i = 0; i < 12; ++i) { v[i] = src[lane + 32 * i]; s += v[i]; }
#pragma unroll
  for (int off = 16; off; off >>= 1) s += __shfl_xor(s, off, 32);
  float mu = s * (1.f / 384.f);
  float var = 0.f;
#pragma unroll
  for (int i = 0; i < 12; ++i) { float d = v[i] - mu; var += d * d; }
#pragma unroll
  for (int off = 16; off; off >>= 1) var += __shfl_xor(var, off, 32);
  float inv = rsqrtf(var * (1.f / 384.f) + 1e-5f);
#pragma unroll
  for (int i = 0; i < 12; ++i)
    dst[lane + 32 * i] = (_Float16)((v[i] - mu) * inv * w[lane + 32 * i] + b[lane + 32 * i]);
}

// ---- QKV GEMM: (65536x384) x (384x1152) -> Q / Kt / V f16 ---------------
__global__ __launch_bounds__(256) void k_gemm_qkv(const _Float16* __restrict__ A,
    const _Float16* __restrict__ Bw, const float* __restrict__ qkv_b,
    _Float16* __restrict__ Q, _Float16* __restrict__ Kt, _Float16* __restrict__ V) {
  __shared__ _Float16 ldsB[2 * LDSB_ELEMS];
  int wid = threadIdx.x >> 5, lane = threadIdx.x & 31;
  int lo = lane & 15, hi = lane >> 4;
  int m0 = blockIdx.y * 128 + (wid >> 1) * 32;
  int nblk = blockIdx.x * 128, n0w = (wid & 1) * 64;
  v8f acc[2][4];
#pragma unroll
  for (int i = 0; i < 2; ++i)
#pragma unroll
    for (int j = 0; j < 4; ++j) acc[i][j] = vzero8();
  gemm_core_lds(A, C_, Bw, 3 * C_, C_, m0, nblk, n0w, ldsB, acc);
#pragma unroll
  for (int nj = 0; nj < 4; ++nj) {
    int n = nblk + n0w + nj * 16 + lo;
    float bias = qkv_b[n];
    int part = n / C_; int nn = n - part * C_;
    int head = nn >> 5, col = nn & 31;
#pragma unroll
    for (int mi = 0; mi < 2; ++mi) {
      int mbase = m0 + mi * 16 + (hi << 3);
      int win = mbase >> 6, r0 = mbase & 63;   // 8 rows stay inside one window
      size_t hb = (size_t)win * NHEAD + head;
      if (part == 0) {                          // Q: scaled, (win,head,row,col)
        _Float16* qp = Q + (hb * 64 + r0) * HD + col;
#pragma unroll
        for (int e = 0; e < 8; ++e)
          qp[(size_t)e * HD] = (_Float16)((acc[mi][nj][e] + bias) * 0.17677669529663687f);
      } else if (part == 1) {                   // K transposed: (win,head,col,row)
        v8h pk;
#pragma unroll
        for (int e = 0; e < 8; ++e) pk[e] = (_Float16)(acc[mi][nj][e] + bias);
        *(v8h*)(Kt + (hb * HD + col) * 64 + r0) = pk;
      } else {                                  // V: (win,head,row,col)
        _Float16* vp = V + (hb * 64 + r0) * HD + col;
#pragma unroll
        for (int e = 0; e < 8; ++e)
          vp[(size_t)e * HD] = (_Float16)(acc[mi][nj][e] + bias);
      }
    }
  }
}

// ---- attention: one wave per (window, head) -----------------------------
__global__ __launch_bounds__(256) void k_attn(const _Float16* __restrict__ Q,
    const _Float16* __restrict__ Kt, const _Float16* __restrict__ V,
    const float* __restrict__ tab4, _Float16* __restrict__ out) {
  __shared__ _Float16 lds[8 * 64 * 64];       // 64 KB: per-wave 64x64 f16 P tile
  int wid = threadIdx.x >> 5, lane = threadIdx.x & 31;
  int lo = lane & 15, hi = lane >> 4;
  int pair = blockIdx.x * 8 + wid;
  int win = pair / NHEAD, head = pair - win * NHEAD;
  int wh = win >> 5, wc = win & 31;
  int type = ((wh == 31) ? 2 : 0) | ((wc == 31) ? 1 : 0);
  const _Float16* q  = Q  + ((size_t)(win * NHEAD + head)) * 64 * HD;
  const _Float16* kt = Kt + ((size_t)(win * NHEAD + head)) * HD * 64;
  const _Float16* v  = V  + ((size_t)(win * NHEAD + head)) * 64 * HD;
  const float* tab = tab4 + ((size_t)(type * NHEAD + head)) * 64 * 64;
  _Float16* plds = lds + wid * 64 * 64;

  // S = q @ k^T  (M=64, N=64, K=32 -> one WMMA K-step, 16 WMMAs)
  v8f s[4][4];
#pragma unroll
  for (int i = 0; i < 4; ++i)
#pragma unroll
    for (int j = 0; j < 4; ++j) s[i][j] = vzero8();
  v16h bf[4];
#pragma unroll
  for (int nj = 0; nj < 4; ++nj) bf[nj] = load_b_frag(kt, 64, 0, nj * 16);
#pragma unroll
  for (int mi = 0; mi < 4; ++mi) {
    v16h a = load_a_frag(q, HD, mi * 16, 0);
#pragma unroll
    for (int nj = 0; nj < 4; ++nj) s[mi][nj] = WMMA_F16(a, bf[nj], s[mi][nj]);
  }
  // + relative-position bias + shift mask + padding mask
#pragma unroll
  for (int mi = 0; mi < 4; ++mi)
#pragma unroll
    for (int nj = 0; nj < 4; ++nj)
#pragma unroll
      for (int e = 0; e < 8; ++e)
        s[mi][nj][e] += tab[(size_t)(mi * 16 + e + (hi << 3)) * 64 + nj * 16 + lo];
  // softmax: rows live in 16-lane halves (C-layout); normalize after P@V
  v8f rs[4];
#pragma unroll
  for (int mi = 0; mi < 4; ++mi) {
    v8f mx = s[mi][0];
#pragma unroll
    for (int nj = 1; nj < 4; ++nj)
#pragma unroll
      for (int e = 0; e < 8; ++e) mx[e] = fmaxf(mx[e], s[mi][nj][e]);
#pragma unroll
    for (int e = 0; e < 8; ++e) {
      float m = mx[e];
#pragma unroll
      for (int off = 1; off < 16; off <<= 1) m = fmaxf(m, __shfl_xor(m, off, 32));
      mx[e] = m;
    }
    v8f sum = vzero8();
#pragma unroll
    for (int nj = 0; nj < 4; ++nj)
#pragma unroll
      for (int e = 0; e < 8; ++e) {
        float p = __expf(s[mi][nj][e] - mx[e]);
        s[mi][nj][e] = p;
        sum[e] += p;
      }
#pragma unroll
    for (int e = 0; e < 8; ++e) {
      float t = sum[e];
#pragma unroll
      for (int off = 1; off < 16; off <<= 1) t += __shfl_xor(t, off, 32);
      rs[mi][e] = 1.0f / t;
    }
    // stash unnormalized P (f16) in LDS; same-wave DS ops are in-order
#pragma unroll
    for (int nj = 0; nj < 4; ++nj)
#pragma unroll
      for (int e = 0; e < 8; ++e)
        plds[(mi * 16 + e + (hi << 3)) * 64 + nj * 16 + lo] = (_Float16)s[mi][nj][e];
  }
  // O = P @ V  (M=64, N=32, K=64 -> 2 K-steps, 16 WMMAs)
  v8f o[4][2];
#pragma unroll
  for (int i = 0; i < 4; ++i) { o[i][0] = vzero8(); o[i][1] = vzero8(); }
#pragma unroll
  for (int kb = 0; kb < 2; ++kb) {
    v16h vb0 = load_b_frag(v, HD, kb * 32, 0);
    v16h vb1 = load_b_frag(v, HD, kb * 32, 16);
#pragma unroll
    for (int mi = 0; mi < 4; ++mi) {
      const _Float16* p = plds + (size_t)(mi * 16 + lo) * 64 + kb * 32 + (hi << 3);
      FragU u; u.h[0] = *(const v8h*)p; u.h[1] = *(const v8h*)(p + 16);
      o[mi][0] = WMMA_F16(u.v, vb0, o[mi][0]);
      o[mi][1] = WMMA_F16(u.v, vb1, o[mi][1]);
    }
  }
  // write the 49 valid rows, normalized, into flat window-order (LTOK x 384)
  size_t tokbase = (size_t)win * NTOK;
#pragma unroll
  for (int mi = 0; mi < 4; ++mi)
#pragma unroll
    for (int e = 0; e < 8; ++e) {
      int m = mi * 16 + e + (hi << 3);
      if (m < NTOK) {
        _Float16* dp = out + (tokbase + m) * C_ + head * HD;
        float r = rs[mi][e];
        dp[lo]      = (_Float16)(o[mi][0][e] * r);
        dp[16 + lo] = (_Float16)(o[mi][1][e] * r);
      }
    }
}

// ---- proj GEMM + window-reverse + unshift + residual -> x1 f32 ----------
__global__ __launch_bounds__(256) void k_gemm_proj(const _Float16* __restrict__ A,
    const _Float16* __restrict__ Bw, const float* __restrict__ proj_b,
    const float* __restrict__ x, float* __restrict__ x1) {
  __shared__ _Float16 ldsB[2 * LDSB_ELEMS];
  int wid = threadIdx.x >> 5, lane = threadIdx.x & 31;
  int lo = lane & 15, hi = lane >> 4;
  int m0 = blockIdx.y * 128 + (wid >> 1) * 32;
  int nblk = blockIdx.x * 128, n0w = (wid & 1) * 64;
  v8f acc[2][4];
#pragma unroll
  for (int i = 0; i < 2; ++i)
#pragma unroll
    for (int j = 0; j < 4; ++j) acc[i][j] = vzero8();
  gemm_core_lds(A, C_, Bw, C_, C_, m0, nblk, n0w, ldsB, acc);
  int ncol[4]; float pb[4];
#pragma unroll
  for (int nj = 0; nj < 4; ++nj) { ncol[nj] = nblk + n0w + nj * 16 + lo; pb[nj] = proj_b[ncol[nj]]; }
#pragma unroll
  for (int mi = 0; mi < 2; ++mi)
#pragma unroll
    for (int e = 0; e < 8; ++e) {
      int i = m0 + mi * 16 + (hi << 3) + e;     // window-order token
      int win = i / NTOK; int t = i - win * NTOK;
      int wh = win >> 5, wc = win & 31;
      int r = t / 7, c = t - r * 7;
      size_t pos = (size_t)((wh * 7 + r + 3) % 224) * 224 + (size_t)((wc * 7 + c + 3) % 224);
      const float* xs = x + pos * C_;
      float* xd = x1 + pos * C_;
#pragma unroll
      for (int nj = 0; nj < 4; ++nj)
        xd[ncol[nj]] = xs[ncol[nj]] + acc[mi][nj][e] + pb[nj];
    }
}

// ---- fc1 GEMM + exact GELU -> h1 f16 ------------------------------------
__global__ __launch_bounds__(256) void k_gemm_fc1(const _Float16* __restrict__ A,
    const _Float16* __restrict__ Bw, const float* __restrict__ fc1_b,
    _Float16* __restrict__ h1) {
  __shared__ _Float16 ldsB[2 * LDSB_ELEMS];
  int wid = threadIdx.x >> 5, lane = threadIdx.x & 31;
  int lo = lane & 15, hi = lane >> 4;
  int m0 = blockIdx.y * 128 + (wid >> 1) * 32;
  int nblk = blockIdx.x * 128, n0w = (wid & 1) * 64;
  v8f acc[2][4];
#pragma unroll
  for (int i = 0; i < 2; ++i)
#pragma unroll
    for (int j = 0; j < 4; ++j) acc[i][j] = vzero8();
  gemm_core_lds(A, C_, Bw, CM_, C_, m0, nblk, n0w, ldsB, acc);
#pragma unroll
  for (int nj = 0; nj < 4; ++nj) {
    int n = nblk + n0w + nj * 16 + lo;
    float bias = fc1_b[n];
#pragma unroll
    for (int mi = 0; mi < 2; ++mi) {
      int mbase = m0 + mi * 16 + (hi << 3);
#pragma unroll
      for (int e = 0; e < 8; ++e) {
        float t = acc[mi][nj][e] + bias;
        float g = 0.5f * t * (1.f + erff(t * 0.70710678118654752f));
        h1[(size_t)(mbase + e) * CM_ + n] = (_Float16)g;
      }
    }
  }
}

// ---- fc2 GEMM + residual -> out f32 -------------------------------------
__global__ __launch_bounds__(256) void k_gemm_fc2(const _Float16* __restrict__ A,
    const _Float16* __restrict__ Bw, const float* __restrict__ fc2_b,
    const float* __restrict__ x1, float* __restrict__ outp) {
  __shared__ _Float16 ldsB[2 * LDSB_ELEMS];
  int wid = threadIdx.x >> 5, lane = threadIdx.x & 31;
  int lo = lane & 15, hi = lane >> 4;
  int m0 = blockIdx.y * 128 + (wid >> 1) * 32;
  int nblk = blockIdx.x * 128, n0w = (wid & 1) * 64;
  v8f acc[2][4];
#pragma unroll
  for (int i = 0; i < 2; ++i)
#pragma unroll
    for (int j = 0; j < 4; ++j) acc[i][j] = vzero8();
  gemm_core_lds(A, CM_, Bw, C_, CM_, m0, nblk, n0w, ldsB, acc);
#pragma unroll
  for (int nj = 0; nj < 4; ++nj) {
    int n = nblk + n0w + nj * 16 + lo;
    float bias = fc2_b[n];
#pragma unroll
    for (int mi = 0; mi < 2; ++mi) {
      int mbase = m0 + mi * 16 + (hi << 3);
#pragma unroll
      for (int e = 0; e < 8; ++e) {
        size_t idx = (size_t)(mbase + e) * C_ + n;
        outp[idx] = x1[idx] + acc[mi][nj][e] + bias;
      }
    }
  }
}

extern "C" void kernel_launch(void* const* d_in, const int* in_sizes, int n_in,
                              void* d_out, int out_size, void* d_ws, size_t ws_size,
                              hipStream_t stream) {
  (void)in_sizes; (void)n_in; (void)out_size; (void)ws_size;
  const float* x      = (const float*)d_in[0];
  const float* rpb    = (const float*)d_in[1];
  const float* qkv_w  = (const float*)d_in[2];
  const float* qkv_b  = (const float*)d_in[3];
  const float* proj_w = (const float*)d_in[4];
  const float* proj_b = (const float*)d_in[5];
  const float* fc1_w  = (const float*)d_in[6];
  const float* fc1_b  = (const float*)d_in[7];
  const float* fc2_w  = (const float*)d_in[8];
  const float* fc2_b  = (const float*)d_in[9];
  const float* n1w    = (const float*)d_in[10];
  const float* n1b    = (const float*)d_in[11];
  const float* n2w    = (const float*)d_in[12];
  const float* n2b    = (const float*)d_in[13];
  float* out = (float*)d_out;

  char* ws = (char*)d_ws;
  size_t off = 0;
  auto carve = [&](size_t bytes) -> void* {
    void* p = ws + off;
    off = (off + bytes + 255) & ~(size_t)255;
    return p;
  };
  _Float16* winA    = (_Float16*)carve((size_t)MPAD * C_ * 2);
  _Float16* qkvwt   = (_Float16*)carve((size_t)C_ * 3 * C_ * 2);
  _Float16* projwt  = (_Float16*)carve((size_t)C_ * C_ * 2);
  _Float16* fc1wt   = (_Float16*)carve((size_t)C_ * CM_ * 2);
  _Float16* fc2wt   = (_Float16*)carve((size_t)CM_ * C_ * 2);
  _Float16* Qb      = (_Float16*)carve((size_t)NWIN * NHEAD * 64 * HD * 2);
  _Float16* Ktb     = (_Float16*)carve((size_t)NWIN * NHEAD * 64 * HD * 2);
  _Float16* Vb      = (_Float16*)carve((size_t)NWIN * NHEAD * 64 * HD * 2);
  float*    tab     = (float*)carve((size_t)4 * NHEAD * 64 * 64 * 4);
  _Float16* attnout = (_Float16*)carve((size_t)LTOK * C_ * 2);
  float*    x1      = (float*)carve((size_t)LTOK * C_ * 4);
  _Float16* yb      = (_Float16*)carve((size_t)LTOK * C_ * 2);
  _Float16* h1      = (_Float16*)carve((size_t)LTOK * CM_ * 2);

  // weight prep (K-major f16) + bias/mask table
  k_transpose_w<<<(3 * C_ * C_ + 255) / 256, 256, 0, stream>>>(qkv_w, qkvwt, 3 * C_, C_);
  k_transpose_w<<<(C_ * C_ + 255) / 256, 256, 0, stream>>>(proj_w, projwt, C_, C_);
  k_transpose_w<<<(CM_ * C_ + 255) / 256, 256, 0, stream>>>(fc1_w, fc1wt, CM_, C_);
  k_transpose_w<<<(C_ * CM_ + 255) / 256, 256, 0, stream>>>(fc2_w, fc2wt, C_, CM_);
  k_biastab<<<(4 * NHEAD * 64 * 64 + 255) / 256, 256, 0, stream>>>(rpb, tab);

  // main pipeline
  k_ln1_window<<<MPAD / 8, 256, 0, stream>>>(x, n1w, n1b, winA);
  k_gemm_qkv<<<dim3((3 * C_) / 128, MPAD / 128), 256, 0, stream>>>(winA, qkvwt, qkv_b, Qb, Ktb, Vb);
  k_attn<<<(NWIN * NHEAD) / 8, 256, 0, stream>>>(Qb, Ktb, Vb, tab, attnout);
  k_gemm_proj<<<dim3(C_ / 128, LTOK / 128), 256, 0, stream>>>(attnout, projwt, proj_b, x, x1);
  k_ln2<<<LTOK / 8, 256, 0, stream>>>(x1, n2w, n2b, yb);
  k_gemm_fc1<<<dim3(CM_ / 128, LTOK / 128), 256, 0, stream>>>(yb, fc1wt, fc1_b, h1);
  k_gemm_fc2<<<dim3(C_ / 128, LTOK / 128), 256, 0, stream>>>(h1, fc2wt, fc2_b, x1, out);
}
